// SelfAttentionV2_10393820857171
// MI455X (gfx1250) — matrix operand order repaired and checked by reading, not compiled, not run on hardware
//
#include <hip/hip_runtime.h>

#define HIDDEN 768
#define SEQ    2048
#define BATCH  4
#define NOUT   (3 * HIDDEN)
#define LOG2E  1.4426950408889634f
#define QSCALE 0.03608439182435161f   // 1/sqrt(768)

typedef __attribute__((ext_vector_type(16))) __bf16 v16bf;
typedef __attribute__((ext_vector_type(8)))  __bf16 v8bf;
typedef __attribute__((ext_vector_type(2)))  __bf16 v2bf;
typedef __attribute__((ext_vector_type(8)))  float  v8f;
typedef __attribute__((ext_vector_type(4)))  float  v4f;

union Frag { v16bf v; v8bf h[2]; };

// float -> bf16, round-to-nearest-even (scalar fallback)
static __device__ __forceinline__ __bf16 f2bf(float f) {
    unsigned u = __builtin_bit_cast(unsigned, f);
    unsigned r = u + 0x7FFFu + ((u >> 16) & 1u);
    unsigned short s = (unsigned short)(r >> 16);
    return __builtin_bit_cast(__bf16, s);
}

// 2x float -> packed 2x bf16 (single v_cvt_pk_bf16_f32 when available)
static __device__ __forceinline__ v2bf f2bf2(float x, float y) {
#if __has_builtin(__builtin_amdgcn_cvt_pk_bf16_f32)
    auto t = __builtin_amdgcn_cvt_pk_bf16_f32(x, y);
    return __builtin_bit_cast(v2bf, t);
#else
    v2bf r; r[0] = f2bf(x); r[1] = f2bf(y); return r;
#endif
}

// ---------------------------------------------------------------------------
// Kernel 0: bulk fp32 -> bf16 conversion (memory bound, one pass).
// ---------------------------------------------------------------------------
__global__ void __launch_bounds__(256) cvt_bf16_kernel(
    const float* __restrict__ src, __bf16* __restrict__ dst, int n8)
{
    const int i = blockIdx.x * 256 + threadIdx.x;
    if (i >= n8) return;
    const v4f* s = (const v4f*)src;
    v4f a = s[2 * i], b = s[2 * i + 1];
    v8bf r;
    v2bf* rp = (v2bf*)&r;
    rp[0] = f2bf2(a[0], a[1]);
    rp[1] = f2bf2(a[2], a[3]);
    rp[2] = f2bf2(b[0], b[1]);
    rp[3] = f2bf2(b[2], b[3]);
    *(v8bf*)(dst + (size_t)i * 8) = r;
}

// ---------------------------------------------------------------------------
// Kernel 1: QKV = X*W^T + b   (M=8192, N=2304, K=768), WMMA bf16, f32 accum.
// LDS-tiled: block tile 128x128, BK=32; 8 waves, each wave owns 2x4 WMMA tiles.
// Writes Q (pre-scaled) row-major bf16, K row-major bf16, V^T [B][H][S] bf16.
// ---------------------------------------------------------------------------
__global__ void __launch_bounds__(256) qkv_gemm_kernel(
    const __bf16* __restrict__ Xb, const __bf16* __restrict__ Wb,
    const float* __restrict__ bias,
    __bf16* __restrict__ Qs, __bf16* __restrict__ Kbuf, __bf16* __restrict__ VT)
{
    __shared__ __bf16 shA[128][40];   // 128 x 32 panel, +8 halves row pad
    __shared__ __bf16 shB[128][40];

    const int w    = threadIdx.x >> 5;
    const int lane = threadIdx.x & 31;
    const int lr   = lane & 15;
    const int kh   = lane >> 4;
    const int wm   = w >> 1;          // 0..3 -> m-tiles 2*wm, 2*wm+1
    const int wn   = w & 1;           // 0..1 -> n-tiles 4*wn .. 4*wn+3
    const int gm0  = blockIdx.x * 128;
    const int gn0  = blockIdx.y * 128;

    const int c0 = threadIdx.x, c1 = threadIdx.x + 256;
    const int r0 = c0 >> 2, p0 = (c0 & 3) * 8;
    const int r1 = c1 >> 2, p1 = (c1 & 3) * 8;

    v8f acc[2][4] = {};

    for (int k0 = 0; k0 < HIDDEN; k0 += 32) {
        __syncthreads();
        // cooperative panel loads: 128x32 bf16 each, 2 x b128 per thread per panel
        *(v8bf*)&shA[r0][p0] = *(const v8bf*)(Xb + (size_t)(gm0 + r0) * HIDDEN + k0 + p0);
        *(v8bf*)&shA[r1][p1] = *(const v8bf*)(Xb + (size_t)(gm0 + r1) * HIDDEN + k0 + p1);
        *(v8bf*)&shB[r0][p0] = *(const v8bf*)(Wb + (size_t)(gn0 + r0) * HIDDEN + k0 + p0);
        *(v8bf*)&shB[r1][p1] = *(const v8bf*)(Wb + (size_t)(gn0 + r1) * HIDDEN + k0 + p1);
        __syncthreads();

        Frag a[2], b[4];
        #pragma unroll
        for (int im = 0; im < 2; ++im) {
            const int row = (2 * wm + im) * 16 + lr;
            a[im].h[0] = *(const v8bf*)&shA[row][kh * 8];
            a[im].h[1] = *(const v8bf*)&shA[row][16 + kh * 8];
        }
        #pragma unroll
        for (int in = 0; in < 4; ++in) {
            const int row = (4 * wn + in) * 16 + lr;
            b[in].h[0] = *(const v8bf*)&shB[row][kh * 8];
            b[in].h[1] = *(const v8bf*)&shB[row][16 + kh * 8];
        }
        #pragma unroll
        for (int im = 0; im < 2; ++im)
            #pragma unroll
            for (int in = 0; in < 4; ++in)
                acc[im][in] = __builtin_amdgcn_wmma_f32_16x16x32_bf16(
                    false, a[im].v, false, b[in].v, (short)0, acc[im][in], false, false);
    }

    // epilogue: add bias, route to Q (scaled) / K / V^T
    #pragma unroll
    for (int in = 0; in < 4; ++in) {
        const int n_t = gn0 + (4 * wn + in) * 16;   // 128 | 768 -> no straddle
        const int sec = n_t / HIDDEN;
        const int h   = (n_t - sec * HIDDEN) + lr;
        const float bn = bias[n_t + lr];
        #pragma unroll
        for (int im = 0; im < 2; ++im) {
            #pragma unroll
            for (int r = 0; r < 8; ++r) {
                const int mrow = gm0 + (2 * wm + im) * 16 + r + 8 * kh;
                const float v = acc[im][in][r] + bn;
                if (sec == 0) {
                    Qs[(size_t)mrow * HIDDEN + h] = f2bf(v * QSCALE);
                } else if (sec == 1) {
                    Kbuf[(size_t)mrow * HIDDEN + h] = f2bf(v);
                } else {
                    const int bb = mrow >> 11, ss = mrow & (SEQ - 1);
                    VT[((size_t)bb * HIDDEN + h) * SEQ + ss] = f2bf(v);
                }
            }
        }
    }
}

// ---------------------------------------------------------------------------
// Kernel 2: flash attention. One block (8 waves) per (batch, 16-query block).
// ---------------------------------------------------------------------------
__global__ void __launch_bounds__(256) attn_kernel(
    const __bf16* __restrict__ Qs, const __bf16* __restrict__ Kbuf,
    const __bf16* __restrict__ VT, float* __restrict__ out)
{
    __shared__ __bf16 shQ[16][HIDDEN + 8];
    __shared__ float  shS[16][128 + 4];
    __shared__ __bf16 shP[16][128 + 8];
    __shared__ float  shC[8][16];

    const int w    = threadIdx.x >> 5;
    const int lane = threadIdx.x & 31;
    const int lr   = lane & 15;
    const int kh   = lane >> 4;

    const int batch = blockIdx.x >> 7;
    const int q0    = (blockIdx.x & 127) * 16;
    const int hb    = w * 96;

    for (int c = threadIdx.x; c < 16 * (HIDDEN / 8); c += 256) {
        const int row = c / (HIDDEN / 8);
        const int col = (c % (HIDDEN / 8)) * 8;
        *(v8bf*)&shQ[row][col] =
            *(const v8bf*)(Qs + (size_t)(batch * SEQ + q0 + row) * HIDDEN + col);
    }
    __syncthreads();

    v8f   accO[6] = {};
    float m_run = -__builtin_inff();
    float l_run = 0.f;

    for (int sb = 0; sb < SEQ / 128; ++sb) {
        const int s0 = sb * 128;

        // ---- scores strip: columns [16w, 16w+16) ----
        v8f accS = {};
        const __bf16* krow =
            Kbuf + (size_t)(batch * SEQ + s0 + w * 16 + lr) * HIDDEN;
        #pragma unroll 4
        for (int k0 = 0; k0 < HIDDEN; k0 += 32) {
            Frag a, b;
            a.h[0] = *(const v8bf*)&shQ[lr][k0 + kh * 8];
            a.h[1] = *(const v8bf*)&shQ[lr][k0 + 16 + kh * 8];
            b.h[0] = *(const v8bf*)(krow + k0 + kh * 8);
            b.h[1] = *(const v8bf*)(krow + k0 + 16 + kh * 8);
            accS = __builtin_amdgcn_wmma_f32_16x16x32_bf16(false, a.v, false, b.v,
                                                           (short)0, accS, false, false);
        }
        #pragma unroll
        for (int r = 0; r < 8; ++r) shS[r + 8 * kh][w * 16 + lr] = accS[r];
        __syncthreads();

        // ---- online softmax: lane handles row lr, cols [kh*64, kh*64+64) ----
        const int cb = kh * 64;
        float mx = -3.0e38f;
        #pragma unroll
        for (int j = 0; j < 16; ++j) {
            v4f s = *(v4f*)&shS[lr][cb + 4 * j];
            mx = fmaxf(mx, fmaxf(fmaxf(s[0], s[1]), fmaxf(s[2], s[3])));
        }
        mx = fmaxf(mx, __shfl_xor(mx, 16, 32));
        const float m_new = fmaxf(m_run, mx);
        const float corr  = exp2f((m_run - m_new) * LOG2E);

        float sum = 0.f;
        #pragma unroll
        for (int j = 0; j < 16; ++j) {
            const int col = cb + 4 * j;
            v4f s = *(v4f*)&shS[lr][col];
            const float e0 = exp2f((s[0] - m_new) * LOG2E);
            const float e1 = exp2f((s[1] - m_new) * LOG2E);
            const float e2 = exp2f((s[2] - m_new) * LOG2E);
            const float e3 = exp2f((s[3] - m_new) * LOG2E);
            sum += (e0 + e1) + (e2 + e3);
            if ((col >> 4) == w) {   // this wave's 16-col strip: packed 32-bit stores
                *(unsigned*)&shP[lr][col]     = __builtin_bit_cast(unsigned, f2bf2(e0, e1));
                *(unsigned*)&shP[lr][col + 2] = __builtin_bit_cast(unsigned, f2bf2(e2, e3));
            }
        }
        sum += __shfl_xor(sum, 16, 32);
        l_run = l_run * corr + sum;
        m_run = m_new;
        if (lane < 16) shC[w][lr] = corr;
        __syncthreads();

        // ---- rescale O, then O += P x V ----
        float cr[8];
        #pragma unroll
        for (int r = 0; r < 8; ++r) cr[r] = shC[w][r + 8 * kh];
        #pragma unroll
        for (int t = 0; t < 6; ++t)
            #pragma unroll
            for (int r = 0; r < 8; ++r) accO[t][r] *= cr[r];

        const __bf16* vbase = VT + (size_t)batch * HIDDEN * SEQ + s0;
        #pragma unroll
        for (int kb = 0; kb < 4; ++kb) {
            Frag a;
            a.h[0] = *(const v8bf*)&shP[lr][kb * 32 + kh * 8];
            a.h[1] = *(const v8bf*)&shP[lr][kb * 32 + 16 + kh * 8];
            #pragma unroll
            for (int t = 0; t < 6; ++t) {
                const __bf16* vp = vbase + (size_t)(hb + t * 16 + lr) * SEQ + kb * 32 + kh * 8;
                Frag b;
                b.h[0] = *(const v8bf*)vp;
                b.h[1] = *(const v8bf*)(vp + 16);
                accO[t] = __builtin_amdgcn_wmma_f32_16x16x32_bf16(false, a.v, false, b.v,
                                                                  (short)0, accO[t], false, false);
            }
        }
    }

    // ---- finalize ----
    if (lane < 16) shC[w][lr] = l_run;
    __syncthreads();
    float rl[8];
    #pragma unroll
    for (int r = 0; r < 8; ++r) rl[r] = 1.0f / shC[w][r + 8 * kh];

    #pragma unroll
    for (int t = 0; t < 6; ++t)
        #pragma unroll
        for (int r = 0; r < 8; ++r)
            out[(size_t)(batch * SEQ + q0 + r + 8 * kh) * HIDDEN + hb + t * 16 + lr] =
                accO[t][r] * rl[r];
}

// ---------------------------------------------------------------------------
extern "C" void kernel_launch(void* const* d_in, const int* in_sizes, int n_in,
                              void* d_out, int out_size, void* d_ws, size_t ws_size,
                              hipStream_t stream)
{
    (void)in_sizes; (void)n_in; (void)out_size; (void)ws_size;
    const float* X = (const float*)d_in[0];
    const float* W = (const float*)d_in[1];
    const float* b = (const float*)d_in[2];
    float* out = (float*)d_out;

    // workspace layout (bf16): Q | K | V^T | Xb | Wb   (~54 MB total)
    const size_t NQK = (size_t)BATCH * SEQ * HIDDEN;   // 6291456
    __bf16* Qs   = (__bf16*)d_ws;
    __bf16* Kbuf = Qs + NQK;
    __bf16* VT   = Kbuf + NQK;
    __bf16* Xb   = VT + NQK;
    __bf16* Wb   = Xb + NQK;

    const int nX8 = (int)(NQK / 8);                    // 786432
    const int nW8 = (NOUT * HIDDEN) / 8;               // 221184

    cvt_bf16_kernel<<<dim3(nX8 / 256), dim3(256), 0, stream>>>(X, Xb, nX8);
    cvt_bf16_kernel<<<dim3(nW8 / 256), dim3(256), 0, stream>>>(W, Wb, nW8);
    // 64 x 18 blocks of 128x128 output tile
    qkv_gemm_kernel<<<dim3(64, 18), dim3(256), 0, stream>>>(Xb, Wb, b, Qs, Kbuf, VT);
    // 4 batches x 128 query blocks
    attn_kernel<<<dim3(BATCH * (SEQ / 16)), dim3(256), 0, stream>>>(Qs, Kbuf, VT, out);
}
